// Net_23673859735759
// MI455X (gfx1250) — matrix operand order repaired and checked by reading, not compiled, not run on hardware
//
#include <hip/hip_runtime.h>
#include <hip/hip_bf16.h>

typedef __attribute__((ext_vector_type(16))) _Float16 v16h;
typedef __attribute__((ext_vector_type(8)))  _Float16 v8h;
typedef __attribute__((ext_vector_type(8)))  float    v8f;

#define WMMA_F16(a, b, c) \
  __builtin_amdgcn_wmma_f32_16x16x32_f16(false, (a), false, (b), (short)0, (c), false, false)

// ---------------------------------------------------------------------------
// helpers
// ---------------------------------------------------------------------------
__device__ __forceinline__ v16h cat8(v8h lo, v8h hi) {
  return __builtin_shufflevector(lo, hi, 0, 1, 2, 3, 4, 5, 6, 7, 8, 9, 10, 11, 12, 13, 14, 15);
}

// load 8 contiguous f32 (32B, b128 x2) and convert to 8 f16
__device__ __forceinline__ v8h cvt8(const float* __restrict__ p) {
  v8f x = *(const v8f*)p;
  v8h r;
#pragma unroll
  for (int i = 0; i < 8; ++i) r[i] = (_Float16)x[i];
  return r;
}

__device__ __forceinline__ void atomicMaxF(float* addr, float v) {
  // valid for mixed signs when buffer is initialized to -FLT_MAX-ish
  if (v >= 0.f) atomicMax((int*)addr, __float_as_int(v));
  else          atomicMin((unsigned int*)addr, __float_as_uint(v));
}

__global__ void k_fill(float* __restrict__ p, float v, long n) {
  long i = (long)blockIdx.x * blockDim.x + threadIdx.x;
  if (i < n) p[i] = v;
}

// ---------------------------------------------------------------------------
// Fragment-ready B operand for V_WMMA_F32_16X16X32_F16.
// Consumer reads fragment f = nt*(K/32)+ks as a per-lane v16h at [(f*32+lane)*16].
// Element j of lane must be B[k][c] = W[c][k], k = ks*32 + (j>>3)*16 + (lane>>4)*8 + (j&7),
// c = nt*16 + (lane&15).  W is OD x K row-major (out_features x in_features).
// ---------------------------------------------------------------------------
__global__ void k_wfrag(const float* __restrict__ W, _Float16* __restrict__ Wf, int OD, int K) {
  int i = blockIdx.x * blockDim.x + threadIdx.x;
  if (i >= OD * K) return;
  const int j    = i & 15;
  const int lane = (i >> 4) & 31;
  const int f    = i >> 9;
  const int nks  = K >> 5;
  const int ks   = f % nks;
  const int nt   = f / nks;
  const int k = ks * 32 + (j >> 3) * 16 + (lane >> 4) * 8 + (j & 7);
  const int c = nt * 16 + (lane & 15);
  Wf[i] = (_Float16)W[(size_t)c * K + k];
}

// ---------------------------------------------------------------------------
// initial embeddings
// ---------------------------------------------------------------------------
__global__ __launch_bounds__(64)
void k_node_embed(const float* __restrict__ nf, const float* __restrict__ W,
                  const float* __restrict__ b, float* __restrict__ out,
                  _Float16* __restrict__ outh) {
  const int n = blockIdx.x, d = threadIdx.x;
  float s = b[d];
#pragma unroll
  for (int k = 0; k < 8; ++k) s += nf[(size_t)n * 8 + k] * W[(size_t)d * 8 + k];
  s = fmaxf(s, 0.f);
  out[(size_t)n * 64 + d]  = s;
  outh[(size_t)n * 64 + d] = (_Float16)s;
}

__global__ void k_edge_embed(const float* __restrict__ ea, const float* __restrict__ W,
                             const float* __restrict__ b, float* __restrict__ out, int E) {
  long i = (long)blockIdx.x * blockDim.x + threadIdx.x;
  if (i >= (long)E * 64) return;
  int e = (int)(i >> 6), d = (int)(i & 63);
  out[i] = fmaxf(ea[e] * W[d] + b[d], 0.f);
}

// ---------------------------------------------------------------------------
// self-loop attr (segment mean of edge_emb over col) + degree
// ---------------------------------------------------------------------------
__global__ __launch_bounds__(64)
void k_deg_loop(const float* __restrict__ ee, const int* __restrict__ col,
                float* __restrict__ loop_attr, float* __restrict__ deg) {
  const int e = blockIdx.x, d = threadIdx.x;
  const int c = col[e];
  atomicAdd(&loop_attr[(size_t)c * 64 + d], ee[(size_t)e * 64 + d]);
  if (d == 0) atomicAdd(&deg[c], 1.f);
}

__global__ void k_loop_norm(float* __restrict__ la, const float* __restrict__ deg, int N) {
  long i = (long)blockIdx.x * blockDim.x + threadIdx.x;
  if (i >= (long)N * 64) return;
  la[i] /= fmaxf(deg[i >> 6], 1.f);
}

// ---------------------------------------------------------------------------
// WMMA GEMM: C(M x 256) = A_h(M x 64) @ W^T + bias; one wave per 16 rows.
// A fragment: lane half h: elems 0..7 -> K=8h+j ; 8..15 -> K=16+8h+(j-8)
// B fragments come pre-swizzled (one v16h load per fragment per lane).
// D: elem r -> row 8h+r, col = lane&15.
// ---------------------------------------------------------------------------
__global__ __launch_bounds__(32)
void k_proj(const _Float16* __restrict__ Ah, const _Float16* __restrict__ Wf,
            const float* __restrict__ bias, float* __restrict__ C) {
  const int base = blockIdx.x * 16;
  const int lane = threadIdx.x, half = lane >> 4, mlo = lane & 15;
  const v8h* ap = (const v8h*)(Ah + (size_t)(base + mlo) * 64);
  const v16h* Bf = (const v16h*)Wf;
  v16h a0 = cat8(ap[half],     ap[2 + half]);
  v16h a1 = cat8(ap[4 + half], ap[6 + half]);
#pragma unroll 1
  for (int nt = 0; nt < 16; ++nt) {
    const int c = nt * 16 + mlo;
    v16h b0 = Bf[(nt * 2 + 0) * 32 + lane];
    v16h b1 = Bf[(nt * 2 + 1) * 32 + lane];
    v8f acc = {0.f, 0.f, 0.f, 0.f, 0.f, 0.f, 0.f, 0.f};
    acc = WMMA_F16(a0, b0, acc);
    acc = WMMA_F16(a1, b1, acc);
    const float bv = bias[c];
#pragma unroll
    for (int r = 0; r < 8; ++r)
      C[(size_t)(base + half * 8 + r) * 256 + c] = acc[r] + bv;
  }
}

// ---------------------------------------------------------------------------
// Fused attention-score kernel: per 16-edge tile compute
// alpha[e][h] = att_h . leaky_relu( (ea2 @ We^T)[e,h,:] + xl[row2[e],h,:] + xr[col2[e],h,:] )
// EE (660K x 256) is never materialized. Atomic segment-max into amax.
// ---------------------------------------------------------------------------
__global__ __launch_bounds__(32)
void k_alpha(const float* __restrict__ edge_emb, const float* __restrict__ loop_attr,
             const int* __restrict__ row, const int* __restrict__ col,
             const float* __restrict__ xl, const float* __restrict__ xr,
             const _Float16* __restrict__ WeF, const float* __restrict__ att,
             float* __restrict__ alpha, float* __restrict__ amax, int E) {
  const int base = blockIdx.x * 16;
  const int lane = threadIdx.x, half = lane >> 4, mlo = lane & 15;

  const int arow = base + mlo;
  const float* ap = (arow < E) ? (edge_emb + (size_t)arow * 64)
                               : (loop_attr + (size_t)(arow - E) * 64);
  v16h a0 = cat8(cvt8(ap + 8 * half),      cvt8(ap + 16 + 8 * half));
  v16h a1 = cat8(cvt8(ap + 32 + 8 * half), cvt8(ap + 48 + 8 * half));
  const v16h* Bf = (const v16h*)WeF;

  int rr[8], cc[8];
#pragma unroll
  for (int r = 0; r < 8; ++r) {
    const int e = base + half * 8 + r;
    if (e < E) { rr[r] = row[e]; cc[r] = col[e]; }
    else       { rr[r] = e - E;  cc[r] = e - E; }
  }

#pragma unroll 1
  for (int h = 0; h < 4; ++h) {
    float acc[8];
#pragma unroll
    for (int r = 0; r < 8; ++r) acc[r] = 0.f;
#pragma unroll 1
    for (int dt = 0; dt < 4; ++dt) {
      const int nt = h * 4 + dt;
      const int c  = nt * 16 + mlo;            // global column in [0,256)
      v16h b0 = Bf[(nt * 2 + 0) * 32 + lane];
      v16h b1 = Bf[(nt * 2 + 1) * 32 + lane];
      v8f ee = {0.f, 0.f, 0.f, 0.f, 0.f, 0.f, 0.f, 0.f};
      ee = WMMA_F16(a0, b0, ee);
      ee = WMMA_F16(a1, b1, ee);
      const float av = att[h * 64 + dt * 16 + mlo];
#pragma unroll
      for (int r = 0; r < 8; ++r) {
        float v = ee[r] + xl[(size_t)rr[r] * 256 + c] + xr[(size_t)cc[r] * 256 + c];
        v = (v > 0.f) ? v : 0.2f * v;          // leaky relu
        acc[r] += v * av;
      }
    }
#pragma unroll
    for (int r = 0; r < 8; ++r) {
      float s = acc[r];
      s += __shfl_xor(s, 1, 32);
      s += __shfl_xor(s, 2, 32);
      s += __shfl_xor(s, 4, 32);
      s += __shfl_xor(s, 8, 32);               // sum over 16 columns (same half)
      if (mlo == 0) {
        const int e = base + half * 8 + r;
        alpha[(size_t)e * 4 + h] = s;
        atomicMaxF(&amax[(size_t)cc[r] * 4 + h], s);
      }
    }
  }
}

__global__ void k_exp_den(const float* __restrict__ amax, float* __restrict__ alpha,
                          float* __restrict__ den, const int* __restrict__ col,
                          int E, int E2) {
  long t = (long)blockIdx.x * blockDim.x + threadIdx.x;
  if (t >= (long)E2 * 4) return;
  const int e = (int)(t >> 2), h = (int)(t & 3);
  const int c2 = (e < E) ? col[e] : (e - E);
  const float v = __expf(alpha[t] - amax[(size_t)c2 * 4 + h]);
  alpha[t] = v;
  atomicAdd(&den[(size_t)c2 * 4 + h], v);
}

__global__ __launch_bounds__(256)
void k_agg(const float* __restrict__ alpha, const float* __restrict__ den,
           const float* __restrict__ xl, const int* __restrict__ row,
           const int* __restrict__ col, float* __restrict__ agg, int E) {
  const int e = blockIdx.x, c = threadIdx.x, h = c >> 6;
  const int r2 = (e < E) ? row[e] : (e - E);
  const int c2 = (e < E) ? col[e] : (e - E);
  const float a = alpha[(size_t)e * 4 + h] / den[(size_t)c2 * 4 + h];
  atomicAdd(&agg[(size_t)c2 * 256 + c], a * xl[(size_t)r2 * 256 + c]);
}

// residual + head-mean + layernorm + relu
__global__ __launch_bounds__(64)
void k_node_update(float* __restrict__ node_emb, _Float16* __restrict__ node_h,
                   const float* __restrict__ agg, const float* __restrict__ bias,
                   const float* __restrict__ ln_g, const float* __restrict__ ln_b) {
  __shared__ float red[64];
  const int n = blockIdx.x, d = threadIdx.x;
  const size_t b = (size_t)n * 256;
  const float upd = 0.25f * (agg[b + d] + agg[b + 64 + d] + agg[b + 128 + d] + agg[b + 192 + d]) + bias[d];
  const float x = node_emb[(size_t)n * 64 + d] + upd;
  red[d] = x; __syncthreads();
  for (int s = 32; s > 0; s >>= 1) { if (d < s) red[d] += red[d + s]; __syncthreads(); }
  const float mu = red[0] * (1.f / 64.f); __syncthreads();
  const float dx = x - mu;
  red[d] = dx * dx; __syncthreads();
  for (int s = 32; s > 0; s >>= 1) { if (d < s) red[d] += red[d + s]; __syncthreads(); }
  const float var = red[0] * (1.f / 64.f);
  float y = dx * rsqrtf(var + 1e-5f) * ln_g[d] + ln_b[d];
  y = fmaxf(y, 0.f);
  node_emb[(size_t)n * 64 + d] = y;
  node_h[(size_t)n * 64 + d]  = (_Float16)y;
}

// ---------------------------------------------------------------------------
// Fused edge MLP: h = relu([x_row | x_col | e] @ W1^T + b1); e = relu(e + h@W2^T + b2)
// One wave per 16 edges; hidden transposed through LDS between the two GEMMs.
// ---------------------------------------------------------------------------
__global__ __launch_bounds__(32)
void k_edge_mlp(float* __restrict__ edge_emb, const _Float16* __restrict__ node_h,
                const int* __restrict__ row, const int* __restrict__ col,
                const _Float16* __restrict__ W1F, const float* __restrict__ b1p,
                const _Float16* __restrict__ W2F, const float* __restrict__ b2p) {
  __shared__ __align__(32) _Float16 hbuf[16 * 64];
  const int base = blockIdx.x * 16;
  const int lane = threadIdx.x, half = lane >> 4, mlo = lane & 15;
  const int e  = base + mlo;
  const int rI = row[e], cI = col[e];
  const v8h* nrow = (const v8h*)(node_h + (size_t)rI * 64);
  const v8h* ncol = (const v8h*)(node_h + (size_t)cI * 64);
  const float* erow = edge_emb + (size_t)e * 64;

  v16h a[6];
  a[0] = cat8(nrow[half],     nrow[2 + half]);
  a[1] = cat8(nrow[4 + half], nrow[6 + half]);
  a[2] = cat8(ncol[half],     ncol[2 + half]);
  a[3] = cat8(ncol[4 + half], ncol[6 + half]);
  a[4] = cat8(cvt8(erow + 8 * half),      cvt8(erow + 16 + 8 * half));
  a[5] = cat8(cvt8(erow + 32 + 8 * half), cvt8(erow + 48 + 8 * half));

  const v16h* Bf1 = (const v16h*)W1F;
#pragma unroll 1
  for (int nt = 0; nt < 4; ++nt) {
    const int c = nt * 16 + mlo;
    v8f acc = {0.f, 0.f, 0.f, 0.f, 0.f, 0.f, 0.f, 0.f};
#pragma unroll 1
    for (int ks = 0; ks < 6; ++ks) {
      v16h b = Bf1[(nt * 6 + ks) * 32 + lane];
      acc = WMMA_F16(a[ks], b, acc);
    }
    const float bv = b1p[c];
#pragma unroll
    for (int r = 0; r < 8; ++r)
      hbuf[(half * 8 + r) * 64 + c] = (_Float16)fmaxf(acc[r] + bv, 0.f);
  }
  __syncthreads();

  const v8h* hp = (const v8h*)(hbuf + mlo * 64);
  v16h ha0 = cat8(hp[half],     hp[2 + half]);
  v16h ha1 = cat8(hp[4 + half], hp[6 + half]);
  const v16h* Bf2 = (const v16h*)W2F;
#pragma unroll 1
  for (int nt = 0; nt < 4; ++nt) {
    const int c = nt * 16 + mlo;
    v16h b0 = Bf2[(nt * 2 + 0) * 32 + lane];
    v16h b1 = Bf2[(nt * 2 + 1) * 32 + lane];
    v8f acc = {0.f, 0.f, 0.f, 0.f, 0.f, 0.f, 0.f, 0.f};
    acc = WMMA_F16(ha0, b0, acc);
    acc = WMMA_F16(ha1, b1, acc);
    const float bv = b2p[c];
#pragma unroll
    for (int r = 0; r < 8; ++r) {
      const size_t idx = (size_t)(base + half * 8 + r) * 64 + c;
      const float v = edge_emb[idx] + acc[r] + bv;
      edge_emb[idx] = fmaxf(v, 0.f);
    }
  }
}

// ---------------------------------------------------------------------------
// Fused heuristic head: 64 -> relu 64 -> relu 32 -> sigmoid 1, per 16-edge tile
// ---------------------------------------------------------------------------
__global__ __launch_bounds__(32)
void k_heur(const float* __restrict__ edge_emb,
            const _Float16* __restrict__ W0F, const float* __restrict__ b0p,
            const _Float16* __restrict__ W1F, const float* __restrict__ b1p,
            const float* __restrict__ W2f, const float* __restrict__ b2p,
            float* __restrict__ out) {
  __shared__ __align__(32) _Float16 h1[16 * 64];
  __shared__ float h2[16 * 32];
  const int base = blockIdx.x * 16;
  const int lane = threadIdx.x, half = lane >> 4, mlo = lane & 15;
  const float* ap = edge_emb + (size_t)(base + mlo) * 64;

  v16h a0 = cat8(cvt8(ap + 8 * half),      cvt8(ap + 16 + 8 * half));
  v16h a1 = cat8(cvt8(ap + 32 + 8 * half), cvt8(ap + 48 + 8 * half));
  const v16h* Bf0 = (const v16h*)W0F;
#pragma unroll 1
  for (int nt = 0; nt < 4; ++nt) {
    const int c = nt * 16 + mlo;
    v16h b0 = Bf0[(nt * 2 + 0) * 32 + lane];
    v16h b1 = Bf0[(nt * 2 + 1) * 32 + lane];
    v8f acc = {0.f, 0.f, 0.f, 0.f, 0.f, 0.f, 0.f, 0.f};
    acc = WMMA_F16(a0, b0, acc);
    acc = WMMA_F16(a1, b1, acc);
    const float bv = b0p[c];
#pragma unroll
    for (int r = 0; r < 8; ++r)
      h1[(half * 8 + r) * 64 + c] = (_Float16)fmaxf(acc[r] + bv, 0.f);
  }
  __syncthreads();

  const v8h* hp = (const v8h*)(h1 + mlo * 64);
  v16h ha0 = cat8(hp[half],     hp[2 + half]);
  v16h ha1 = cat8(hp[4 + half], hp[6 + half]);
  const v16h* Bf1 = (const v16h*)W1F;
#pragma unroll 1
  for (int nt = 0; nt < 2; ++nt) {
    const int c = nt * 16 + mlo;
    v16h b0 = Bf1[(nt * 2 + 0) * 32 + lane];
    v16h b1 = Bf1[(nt * 2 + 1) * 32 + lane];
    v8f acc = {0.f, 0.f, 0.f, 0.f, 0.f, 0.f, 0.f, 0.f};
    acc = WMMA_F16(ha0, b0, acc);
    acc = WMMA_F16(ha1, b1, acc);
    const float bv = b1p[c];
#pragma unroll
    for (int r = 0; r < 8; ++r)
      h2[(half * 8 + r) * 32 + c] = fmaxf(acc[r] + bv, 0.f);
  }
  __syncthreads();

  if (lane < 16) {
    float s = b2p[0];
#pragma unroll
    for (int j = 0; j < 32; ++j) s += h2[lane * 32 + j] * W2f[j];
    out[base + lane] = 1.f / (1.f + __expf(-s));
  }
}

// ---------------------------------------------------------------------------
// critic
// ---------------------------------------------------------------------------
__global__ __launch_bounds__(64)
void k_pool(const float* __restrict__ node_emb, float* __restrict__ gsum, int N) {
  const int d = threadIdx.x;
  float s = 0.f;
  for (int n = blockIdx.x; n < N; n += gridDim.x) s += node_emb[(size_t)n * 64 + d];
  atomicAdd(&gsum[d], s);
}

__global__ __launch_bounds__(128)
void k_critic(const float* __restrict__ gsum, const float* __restrict__ cW1,
              const float* __restrict__ cb1, const float* __restrict__ cW2,
              const float* __restrict__ cb2, float* __restrict__ out, float invN) {
  __shared__ float red[128];
  const int j = threadIdx.x;
  float s = cb1[j];
  for (int d = 0; d < 64; ++d) s += gsum[d] * invN * cW1[(size_t)j * 64 + d];
  const float v = s / (1.f + __expf(-s));   // silu
  red[j] = v * cW2[j]; __syncthreads();
  for (int t = 64; t > 0; t >>= 1) { if (j < t) red[j] += red[j + t]; __syncthreads(); }
  if (j == 0) out[0] = red[0] + cb2[0];
}

// ---------------------------------------------------------------------------
// host
// ---------------------------------------------------------------------------
extern "C" void kernel_launch(void* const* d_in, const int* in_sizes, int n_in,
                              void* d_out, int out_size, void* d_ws, size_t ws_size,
                              hipStream_t stream) {
  (void)n_in; (void)out_size; (void)ws_size;
  // Detect flattening order: insertion-order dict vs jax sorted tree_leaves.
  const bool ins = (in_sizes[0] == 160000);   // node_feat (20000*8) first => insertion
  int i_nf, i_ei, i_ea, i_nW, i_nb, i_eW, i_eb;
  int i_hW0, i_hb0, i_hW1, i_hb1, i_hW2, i_hb2, i_cW1, i_cb1, i_cW2, i_cb2;
  int lb[4][13];  // slots: 0=Wl 1=bl 2=Wr 3=br 4=We 5=att 6=bias 7=ln_g 8=ln_b 9=W1 10=b1 11=W2 12=b2
  if (ins) {
    i_nf = 0; i_ei = 1; i_ea = 2; i_nW = 3; i_nb = 4; i_eW = 5; i_eb = 6;
    for (int L = 0; L < 4; ++L) { int b = 7 + L * 13; for (int s = 0; s < 13; ++s) lb[L][s] = b + s; }
    i_hW0 = 59; i_hb0 = 60; i_hW1 = 61; i_hb1 = 62; i_hW2 = 63; i_hb2 = 64;
    i_cW1 = 65; i_cb1 = 66; i_cW2 = 67; i_cb2 = 68;
  } else {
    i_ea = 0; i_ei = 1; i_nf = 2;
    i_cW1 = 3; i_cW2 = 4; i_cb1 = 5; i_cb2 = 6; i_eW = 7; i_eb = 8;
    i_hW0 = 9; i_hb0 = 10; i_hW1 = 11; i_hb1 = 12; i_hW2 = 13; i_hb2 = 14;
    for (int L = 0; L < 4; ++L) {
      int b = 15 + L * 13;  // sorted keys: W1,W2,We,Wl,Wr,att,b1,b2,bias,bl,br,ln_b,ln_g
      lb[L][9] = b + 0;  lb[L][11] = b + 1; lb[L][4] = b + 2;  lb[L][0] = b + 3;
      lb[L][2] = b + 4;  lb[L][5]  = b + 5; lb[L][10] = b + 6; lb[L][12] = b + 7;
      lb[L][6] = b + 8;  lb[L][1]  = b + 9; lb[L][3] = b + 10; lb[L][8] = b + 11;
      lb[L][7] = b + 12;
    }
    i_nW = 67; i_nb = 68;
  }
  auto F = [&](int i) { return (const float*)d_in[i]; };
  const int* ei  = (const int*)d_in[i_ei];
  const int  N   = in_sizes[i_nf] / 8;     // 20000
  const int  E   = in_sizes[i_ea];         // 640000
  const int  E2  = E + N;                  // 660000 (all multiples of 16)
  const int* row = ei;
  const int* col = ei + E;

  char* w = (char*)d_ws;
  auto A = [&](size_t bytes) { char* p = w; w += (bytes + 255) & ~(size_t)255; return p; };
  float*    node_emb  = (float*)A((size_t)N * 64 * 4);
  _Float16* node_h    = (_Float16*)A((size_t)N * 64 * 2);
  float*    edge_emb  = (float*)A((size_t)E * 64 * 4);
  float*    loop_attr = (float*)A((size_t)N * 64 * 4);
  float*    deg       = (float*)A((size_t)N * 4);
  float*    xl        = (float*)A((size_t)N * 256 * 4);
  float*    xr        = (float*)A((size_t)N * 256 * 4);
  float*    alpha     = (float*)A((size_t)E2 * 4 * 4);
  float*    amax      = (float*)A((size_t)N * 4 * 4);
  float*    den       = (float*)A((size_t)N * 4 * 4);
  float*    agg       = (float*)A((size_t)N * 256 * 4);
  float*    gsum      = (float*)A(64 * 4);
  _Float16* WlF  = (_Float16*)A(64 * 256 * 2);
  _Float16* WrF  = (_Float16*)A(64 * 256 * 2);
  _Float16* WeF  = (_Float16*)A(64 * 256 * 2);
  _Float16* W1F  = (_Float16*)A(192 * 64 * 2);
  _Float16* W2F  = (_Float16*)A(64 * 64 * 2);
  _Float16* hW0F = (_Float16*)A(64 * 64 * 2);
  _Float16* hW1F = (_Float16*)A(64 * 32 * 2);

  float* hout = (float*)d_out;

  k_node_embed<<<N, 64, 0, stream>>>(F(i_nf), F(i_nW), F(i_nb), node_emb, node_h);
  {
    long n = (long)E * 64;
    k_edge_embed<<<(int)((n + 255) / 256), 256, 0, stream>>>(F(i_ea), F(i_eW), F(i_eb), edge_emb, E);
  }

  for (int L = 0; L < 4; ++L) {
    k_wfrag<<<(256 * 64 + 255) / 256, 256, 0, stream>>>(F(lb[L][0]), WlF, 256, 64);
    k_wfrag<<<(256 * 64 + 255) / 256, 256, 0, stream>>>(F(lb[L][2]), WrF, 256, 64);
    k_wfrag<<<(256 * 64 + 255) / 256, 256, 0, stream>>>(F(lb[L][4]), WeF, 256, 64);
    k_wfrag<<<(64 * 192 + 255) / 256, 256, 0, stream>>>(F(lb[L][9]), W1F, 64, 192);
    k_wfrag<<<(64 * 64 + 255) / 256, 256, 0, stream>>>(F(lb[L][11]), W2F, 64, 64);

    k_fill<<<(N + 255) / 256, 256, 0, stream>>>(deg, 0.f, N);
    k_fill<<<(int)(((long)N * 64 + 255) / 256), 256, 0, stream>>>(loop_attr, 0.f, (long)N * 64);
    k_fill<<<(N * 4 + 255) / 256, 256, 0, stream>>>(amax, -3.0e38f, (long)N * 4);
    k_fill<<<(N * 4 + 255) / 256, 256, 0, stream>>>(den, 0.f, (long)N * 4);
    k_fill<<<(int)(((long)N * 256 + 255) / 256), 256, 0, stream>>>(agg, 0.f, (long)N * 256);

    k_deg_loop<<<E, 64, 0, stream>>>(edge_emb, col, loop_attr, deg);
    k_loop_norm<<<(int)(((long)N * 64 + 255) / 256), 256, 0, stream>>>(loop_attr, deg, N);

    k_proj<<<N / 16, 32, 0, stream>>>(node_h, WlF, F(lb[L][1]), xl);
    k_proj<<<N / 16, 32, 0, stream>>>(node_h, WrF, F(lb[L][3]), xr);

    k_alpha<<<E2 / 16, 32, 0, stream>>>(edge_emb, loop_attr, row, col, xl, xr, WeF,
                                        F(lb[L][5]), alpha, amax, E);
    k_exp_den<<<(int)(((long)E2 * 4 + 255) / 256), 256, 0, stream>>>(amax, alpha, den, col, E, E2);
    k_agg<<<E2, 256, 0, stream>>>(alpha, den, xl, row, col, agg, E);
    k_node_update<<<N, 64, 0, stream>>>(node_emb, node_h, agg, F(lb[L][6]), F(lb[L][7]), F(lb[L][8]));
    k_edge_mlp<<<E / 16, 32, 0, stream>>>(edge_emb, node_h, row, col, W1F, F(lb[L][10]),
                                          W2F, F(lb[L][12]));
  }

  k_wfrag<<<(64 * 64 + 255) / 256, 256, 0, stream>>>(F(i_hW0), hW0F, 64, 64);
  k_wfrag<<<(32 * 64 + 255) / 256, 256, 0, stream>>>(F(i_hW1), hW1F, 32, 64);
  k_heur<<<E / 16, 32, 0, stream>>>(edge_emb, hW0F, F(i_hb0), hW1F, F(i_hb1),
                                    F(i_hW2), F(i_hb2), hout);

  k_fill<<<1, 64, 0, stream>>>(gsum, 0.f, 64);
  k_pool<<<256, 64, 0, stream>>>(node_emb, gsum, N);
  k_critic<<<1, 128, 0, stream>>>(gsum, F(i_cW1), F(i_cb1), F(i_cW2), F(i_cb2),
                                  hout + E, 1.f / (float)N);
}
